// block_1D_8718783611642
// MI455X (gfx1250) — matrix operand order repaired and checked by reading, not compiled, not run on hardware
//
#include <hip/hip_runtime.h>
#include <math.h>
#include <stdint.h>

// Problem constants (from reference)
#define B_   16
#define L_   2048
#define DM_  512
#define DI_  1024
#define DS_  16
#define DR_  32
#define KC_  4
#define NROW (B_ * L_)   // 32768 rows of "tokens"

typedef __attribute__((ext_vector_type(16))) __bf16          v16bf;
typedef __attribute__((ext_vector_type(8)))  __bf16          bf16x8;
typedef __attribute__((ext_vector_type(8)))  float           v8f;
typedef __attribute__((ext_vector_type(8)))  unsigned short  us8;
typedef __attribute__((ext_vector_type(4)))  unsigned int    uint4v;
typedef __attribute__((ext_vector_type(8)))  int             int8v;
typedef __attribute__((ext_vector_type(4)))  int             int4v;

// ---- Tensor Data Mover availability / arity selection -----------------------
#if defined(__has_builtin)
#if __has_builtin(__builtin_amdgcn_tensor_load_to_lds)
#define HAVE_TDM 1
#if __has_include(<hip/amd_detail/amd_gfx1250_TDM.h>)
// therock-10.0 headers -> 6-arg builtin
#define TDM_LOAD(g0, g1, g2, g3) \
  __builtin_amdgcn_tensor_load_to_lds((g0), (g1), (g2), (g3), (int8v){0,0,0,0,0,0,0,0}, 0)
#else
// ROCm 7.2 clang-22 -> 5-arg builtin
#define TDM_LOAD(g0, g1, g2, g3) \
  __builtin_amdgcn_tensor_load_to_lds((g0), (g1), (g2), (g3), 0)
#endif
#endif
#endif
#ifndef HAVE_TDM
#define HAVE_TDM 0
#endif

__device__ __forceinline__ unsigned short f2bf(float f) {
  unsigned u = __builtin_bit_cast(unsigned, f);
  u += 0x7FFFu + ((u >> 16) & 1u);   // round-to-nearest-even
  return (unsigned short)(u >> 16);
}

// ---------------------------------------------------------------------------
// LayerNorm over DM=512, output bf16 (feeds the W_in WMMA GEMM)
// ---------------------------------------------------------------------------
__global__ __launch_bounds__(256)
void ln_bf16_kernel(const float* __restrict__ x, const float* __restrict__ w,
                    const float* __restrict__ b, unsigned short* __restrict__ xo) {
  int row = blockIdx.x, tid = threadIdx.x;
  const float* xr = x + (size_t)row * DM_;
  float a0 = xr[tid], a1 = xr[tid + 256];
  __shared__ float s1[256], s2[256];
  s1[tid] = a0 + a1;
  s2[tid] = a0 * a0 + a1 * a1;
  __syncthreads();
  for (int s = 128; s > 0; s >>= 1) {
    if (tid < s) { s1[tid] += s1[tid + s]; s2[tid] += s2[tid + s]; }
    __syncthreads();
  }
  float mu  = s1[0] * (1.0f / DM_);
  float var = s2[0] * (1.0f / DM_) - mu * mu;
  float rs  = rsqrtf(var + 1e-6f);
  unsigned short* o = xo + (size_t)row * DM_;
  o[tid]       = f2bf((a0 - mu) * rs * w[tid]       + b[tid]);
  o[tid + 256] = f2bf((a1 - mu) * rs * w[tid + 256] + b[tid + 256]);
}

// ---------------------------------------------------------------------------
// f32 -> bf16 weight conversion
// ---------------------------------------------------------------------------
__global__ void cvt_bf16_kernel(const float* __restrict__ src,
                                unsigned short* __restrict__ dst, int n) {
  int i = blockIdx.x * 256 + threadIdx.x;
  if (i < n) dst[i] = f2bf(src[i]);
}

// ---------------------------------------------------------------------------
// bf16 WMMA GEMM: C(MxN,f32) = A(MxK,bf16) * B(KxN,bf16), fused epilogues.
// Block tile 128x64, 8 waves (4x2), wave tile 32x32 (2x2 v_wmma_f32_16x16x32_bf16).
// A tile staged via Tensor Data Mover (TDM) when available and rows are linear;
// TDM hardware padding reproduces the 80B LDS row stride.
// ---------------------------------------------------------------------------
#define TBM 128
#define TBN 64
#define TBK 32
#define LDS_STR 40   // ushort row stride (80B, 16B-aligned, padded vs 64 banks)

enum { EPI_STORE = 0, EPI_SOFTPLUS = 1, EPI_XDBL = 2, EPI_FINAL = 3 };

__global__ __launch_bounds__(256)
void gemm_bf16_wmma(const unsigned short* __restrict__ A,
                    const unsigned short* __restrict__ Bw,
                    float* __restrict__ C,
                    int M, int N, int K,
                    int mode, int aremap,
                    const float* __restrict__ bias,        // EPI_SOFTPLUS
                    unsigned short* __restrict__ aux,      // EPI_XDBL: bf16 copy of first DR cols
                    const float* __restrict__ y1,          // EPI_FINAL
                    const float* __restrict__ resid,       // EPI_FINAL
                    float* __restrict__ outp) {            // EPI_FINAL
  __shared__ unsigned short As[TBM][LDS_STR];   // [m][k]
  __shared__ unsigned short Bs[TBN][LDS_STR];   // transposed: [n][k]

  int tid = threadIdx.x;
  int lane = tid & 31, w = tid >> 5;
  int wm = w >> 1, wn = w & 1;          // 4x2 wave grid
  int m0 = blockIdx.y * TBM;
  int n0 = blockIdx.x * TBN;

  // --- A global row (optional time-reversal remap for the backward pass) ---
  int lr  = tid >> 1;                   // 0..127
  int lcb = (tid & 1) << 4;             // 0 / 16
  int grow = m0 + lr;
  if (aremap) {
    int bb = grow / L_, tt = grow - bb * L_;
    int rt = (tt == L_ - 1) ? (L_ - 1) : (L_ - 2 - tt);
    grow = bb * L_ + rt;
  }
  const unsigned short* arow = A + (size_t)grow * K;

  int brn = tid >> 3;                   // 0..31 : k within tile
  int bcn = (tid & 7) << 3;             // n offset within tile

  v8f acc[2][2];
  #pragma unroll
  for (int i = 0; i < 2; i++)
    #pragma unroll
    for (int j = 0; j < 2; j++)
      acc[i][j] = (v8f){0.f,0.f,0.f,0.f,0.f,0.f,0.f,0.f};

  int am    = lane & 15;
  int abase = (lane >= 16) ? 8  : 0;    // A-matrix 16-bit layout: hi lanes at K+8
  int bbase = (lane >= 16) ? 8  : 0;    // B-matrix 16-bit layout: hi lanes at K+8

#if HAVE_TDM
  bool use_tdm = (aremap == 0);         // linear rows only; wave-uniform
  unsigned ldsA = (unsigned)(uintptr_t)(&As[0][0]);
  unsigned long long gabase =
      (unsigned long long)(uintptr_t)A + (unsigned long long)m0 * (unsigned)K * 2ull;
#endif

  for (int k0 = 0; k0 < K; k0 += TBK) {
#if HAVE_TDM
    if (use_tdm) {
      if (tid < 32) {  // wave 0 issues one DMA for the whole 128x32 A tile
        unsigned long long ga = gabase + (unsigned long long)k0 * 2ull;
        uint4v g0;
        g0.x = 1u;                                        // count=1 (valid D#)
        g0.y = ldsA;                                      // lds_addr (bytes)
        g0.z = (unsigned)(ga & 0xffffffffull);            // global_addr[31:0]
        g0.w = (unsigned)((ga >> 32) & 0x1ffffffull)      // global_addr[56:32]
               | (2u << 30);                              // type=2 ("image")
        int8v g1;
        g1[0] = (int)((1u << 16)      // data_size = 2 bytes
                    | (1u << 20)      // pad_enable
                    | (3u << 22)      // pad_interval: 16 DWORDs (64B = one tile row)
                    | (3u << 25));    // pad_amount: 4 DWORDs (16B) -> 80B LDS stride
        g1[1] = (int)(((unsigned)K & 0xffffu) << 16);     // tensor_dim0[15:0]
        g1[2] = (int)((((unsigned)K >> 16) & 0xffffu)     // tensor_dim0[31:16]
                    | (((unsigned)M & 0xffffu) << 16));   // tensor_dim1[15:0]
        g1[3] = (int)((((unsigned)M >> 16) & 0xffffu)     // tensor_dim1[31:16]
                    | ((unsigned)TBK << 16));             // tile_dim0 = 32
        g1[4] = (int)TBM;                                 // tile_dim1 = 128 (tile_dim2=0)
        g1[5] = (int)K;                                   // tensor_dim0_stride[31:0]
        g1[6] = 0;                                        // stride0 hi, stride1 lo
        g1[7] = 0;
        int4v gz = {0, 0, 0, 0};
        TDM_LOAD(g0, g1, gz, gz);
        __builtin_amdgcn_s_wait_tensorcnt(0);
      }
    } else
#endif
    {
      // manual staging: each thread 16 contiguous bf16 = 2x b128
      us8 av0 = *(const us8*)(arow + k0 + lcb);
      us8 av1 = *(const us8*)(arow + k0 + lcb + 8);
      *(us8*)&As[lr][lcb]     = av0;
      *(us8*)&As[lr][lcb + 8] = av1;
    }
    // stage B tile, transposed into [n][k]
    us8 bv = *(const us8*)(Bw + (size_t)(k0 + brn) * N + n0 + bcn);
    #pragma unroll
    for (int i = 0; i < 8; i++) Bs[bcn + i][brn] = bv[i];
    __syncthreads();

    v16bf afrag[2], bfrag[2];
    #pragma unroll
    for (int tm = 0; tm < 2; tm++) {
      const unsigned short* ap = &As[wm * 32 + tm * 16 + am][0];
      bf16x8 lo = *(const bf16x8*)(ap + abase);        // K = abase .. abase+7
      bf16x8 hi = *(const bf16x8*)(ap + abase + 16);   // K = abase+16 .. abase+23
      afrag[tm] = __builtin_shufflevector(lo, hi, 0,1,2,3,4,5,6,7,8,9,10,11,12,13,14,15);
    }
    #pragma unroll
    for (int tn = 0; tn < 2; tn++) {
      const unsigned short* bp = &Bs[wn * 32 + tn * 16 + am][0];
      bf16x8 lo = *(const bf16x8*)(bp + bbase);        // K = bbase .. bbase+7
      bf16x8 hi = *(const bf16x8*)(bp + bbase + 16);   // K = bbase+16 .. bbase+23
      bfrag[tn] = __builtin_shufflevector(lo, hi, 0,1,2,3,4,5,6,7,8,9,10,11,12,13,14,15);
    }
    #pragma unroll
    for (int tm = 0; tm < 2; tm++)
      #pragma unroll
      for (int tn = 0; tn < 2; tn++)
        acc[tm][tn] = __builtin_amdgcn_wmma_f32_16x16x32_bf16(
            false, afrag[tm], false, bfrag[tn], (short)0, acc[tm][tn], false, false);
    __syncthreads();
  }

  // --- epilogue (C/D layout: lane 0-15 -> M=r, lane 16-31 -> M=r+8; N=lane%16) ---
  int lhalf8 = (lane >= 16) ? 8 : 0;
  #pragma unroll
  for (int tm = 0; tm < 2; tm++) {
    #pragma unroll
    for (int tn = 0; tn < 2; tn++) {
      int n = n0 + wn * 32 + tn * 16 + am;
      #pragma unroll
      for (int r = 0; r < 8; r++) {
        int m = m0 + wm * 32 + tm * 16 + r + lhalf8;
        float v = acc[tm][tn][r];
        size_t idx = (size_t)m * N + n;
        if (mode == EPI_STORE) {
          C[idx] = v;
        } else if (mode == EPI_SOFTPLUS) {
          float t = v + bias[n];
          C[idx] = (t > 20.0f) ? t : log1pf(__expf(t));
        } else if (mode == EPI_XDBL) {
          C[idx] = v;
          if (n < DR_) aux[(size_t)m * DR_ + n] = f2bf(v);
        } else { // EPI_FINAL: out[b, rev(t)] = 0.5*(y3 + y1[b,rev(t)]) + input
          int bb = m / L_, tt = m - bb * L_;
          int rt = (tt == L_ - 1) ? (L_ - 1) : (L_ - 2 - tt);
          size_t o = ((size_t)(bb * L_ + rt)) * N + n;
          outp[o] = 0.5f * (v + y1[o]) + resid[o];
        }
      }
    }
  }
}

// ---------------------------------------------------------------------------
// Depthwise causal conv (K=4) + SiLU. Reads xp half of xz; writes f32 + bf16.
// ---------------------------------------------------------------------------
__global__ __launch_bounds__(256)
void conv_silu_kernel(const float* __restrict__ xz,
                      const float* __restrict__ cw, const float* __restrict__ cb,
                      float* __restrict__ xp, unsigned short* __restrict__ xpb) {
  size_t idx = (size_t)blockIdx.x * 256 + threadIdx.x;   // over B*L*DI
  int d = (int)(idx % DI_);
  size_t bt = idx / DI_;
  int t = (int)(bt % L_);
  float acc = cb[d];
  const float* base = xz + bt * (size_t)(2 * DI_) + d;
  #pragma unroll
  for (int k = 0; k < KC_; k++) {
    int tt = t - (KC_ - 1) + k;
    if (tt >= 0)
      acc += cw[d * KC_ + k] * base[(ptrdiff_t)(k - (KC_ - 1)) * (2 * DI_)];
  }
  float s = acc / (1.0f + __expf(-acc));
  xp[idx]  = s;
  xpb[idx] = f2bf(s);
}

// ---------------------------------------------------------------------------
// Selective scan. One thread per (b, d) channel; h[16] state in VGPRs.
// Fuses +xp*D, SiLU(z) gating; writes bf16 for the W_out WMMA GEMM.
// ---------------------------------------------------------------------------
__global__ __launch_bounds__(256)
void scan_kernel(const float* __restrict__ dt,  const float* __restrict__ xp,
                 const float* __restrict__ xdbl, const float* __restrict__ xz,
                 const float* __restrict__ A_log, const float* __restrict__ Dp,
                 unsigned short* __restrict__ yg) {
  int gid = blockIdx.x * 256 + threadIdx.x;   // 0 .. B*DI-1
  int b = gid / DI_;
  int d = gid - b * DI_;
  float Arow[DS_];
  #pragma unroll
  for (int s = 0; s < DS_; s++) Arow[s] = -__expf(A_log[d * DS_ + s]);
  float Dd = Dp[d];
  float h[DS_];
  #pragma unroll
  for (int s = 0; s < DS_; s++) h[s] = 0.f;
  size_t rbase = (size_t)b * L_;
  for (int t = 0; t < L_; t++) {
    size_t row = rbase + t;
    float dtv = dt[row * DI_ + d];
    float xv  = xp[row * DI_ + d];
    float zv  = xz[row * (size_t)(2 * DI_) + DI_ + d];
    const float* bc = xdbl + row * 64 + DR_;      // B (16) then C (16)
    if (t + 1 < L_) __builtin_prefetch(xdbl + (row + 1) * 64 + DR_, 0, 1);
    float dx = dtv * xv;
    float y = 0.f;
    #pragma unroll
    for (int s = 0; s < DS_; s++) {
      float dA = __expf(dtv * Arow[s]);
      h[s] = dA * h[s] + dx * bc[s];
      y += h[s] * bc[DS_ + s];
    }
    y += xv * Dd;
    float g = y * (zv / (1.0f + __expf(-zv)));
    yg[row * DI_ + d] = f2bf(g);
  }
}

// ---------------------------------------------------------------------------
extern "C" void kernel_launch(void* const* d_in, const int* in_sizes, int n_in,
                              void* d_out, int out_size, void* d_ws, size_t ws_size,
                              hipStream_t stream) {
  const float* input  = (const float*)d_in[0];
  const float* ln_w   = (const float*)d_in[1];
  const float* ln_b   = (const float*)d_in[2];
  const float* W_in   = (const float*)d_in[3];
  const float* conv_w = (const float*)d_in[4];
  const float* conv_b = (const float*)d_in[5];
  const float* W_x    = (const float*)d_in[6];
  const float* W_dt   = (const float*)d_in[7];
  const float* b_dt   = (const float*)d_in[8];
  const float* A_log  = (const float*)d_in[9];
  const float* Dp     = (const float*)d_in[10];
  const float* W_out  = (const float*)d_in[11];
  float* out = (float*)d_out;

  char* ws = (char*)d_ws;
  size_t off = 0;
  auto alloc = [&](size_t bytes) -> char* {
    char* p = ws + off;
    off += (bytes + 255) & ~(size_t)255;
    return p;
  };
  unsigned short* x_ln   = (unsigned short*)alloc((size_t)NROW * DM_ * 2);
  unsigned short* Win_b  = (unsigned short*)alloc((size_t)DM_ * 2 * DI_ * 2);
  unsigned short* Wx_b   = (unsigned short*)alloc((size_t)DI_ * 64 * 2);
  unsigned short* Wdt_b  = (unsigned short*)alloc((size_t)DR_ * DI_ * 2);
  unsigned short* Wout_b = (unsigned short*)alloc((size_t)DI_ * DM_ * 2);
  float*          xz     = (float*)         alloc((size_t)NROW * 2 * DI_ * 4);
  float*          xp     = (float*)         alloc((size_t)NROW * DI_ * 4);
  unsigned short* xp_b   = (unsigned short*)alloc((size_t)NROW * DI_ * 2);
  float*          xdbl   = (float*)         alloc((size_t)NROW * 64 * 4);
  unsigned short* dtr_b  = (unsigned short*)alloc((size_t)NROW * DR_ * 2);
  float*          dtf    = (float*)         alloc((size_t)NROW * DI_ * 4);
  unsigned short* yg_b   = (unsigned short*)alloc((size_t)NROW * DI_ * 2);
  float*          y1     = (float*)         alloc((size_t)NROW * DM_ * 4);

  // Weights -> bf16 (once per launch; small)
  cvt_bf16_kernel<<<(DM_ * 2 * DI_ + 255) / 256, 256, 0, stream>>>(W_in,  Win_b,  DM_ * 2 * DI_);
  cvt_bf16_kernel<<<(DI_ * 64    + 255) / 256, 256, 0, stream>>>(W_x,   Wx_b,   DI_ * 64);
  cvt_bf16_kernel<<<(DR_ * DI_   + 255) / 256, 256, 0, stream>>>(W_dt,  Wdt_b,  DR_ * DI_);
  cvt_bf16_kernel<<<(DI_ * DM_   + 255) / 256, 256, 0, stream>>>(W_out, Wout_b, DI_ * DM_);

  ln_bf16_kernel<<<NROW, 256, 0, stream>>>(input, ln_w, ln_b, x_ln);

  for (int pass = 0; pass < 2; pass++) {
    // xz = x_ln @ W_in   (pass 1: rows read through rev(t) remap)
    gemm_bf16_wmma<<<dim3((2 * DI_) / TBN, NROW / TBM), 256, 0, stream>>>(
        x_ln, Win_b, xz, NROW, 2 * DI_, DM_, EPI_STORE, pass,
        nullptr, nullptr, nullptr, nullptr, nullptr);
    // depthwise conv + SiLU
    conv_silu_kernel<<<(unsigned)((size_t)NROW * DI_ / 256), 256, 0, stream>>>(
        xz, conv_w, conv_b, xp, xp_b);
    // x_dbl = xp @ W_x (also emit bf16 copy of the dt-rank columns)
    gemm_bf16_wmma<<<dim3(64 / TBN, NROW / TBM), 256, 0, stream>>>(
        xp_b, Wx_b, xdbl, NROW, 64, DI_, EPI_XDBL, 0,
        nullptr, dtr_b, nullptr, nullptr, nullptr);
    // dt = softplus(dtr @ W_dt + b_dt)
    gemm_bf16_wmma<<<dim3(DI_ / TBN, NROW / TBM), 256, 0, stream>>>(
        dtr_b, Wdt_b, dtf, NROW, DI_, DR_, EPI_SOFTPLUS, 0,
        b_dt, nullptr, nullptr, nullptr, nullptr);
    // selective scan + D skip + SiLU(z) gate -> bf16
    scan_kernel<<<(B_ * DI_) / 256, 256, 0, stream>>>(
        dtf, xp, xdbl, xz, A_log, Dp, yg_b);
    // y @ W_out
    if (pass == 0) {
      gemm_bf16_wmma<<<dim3(DM_ / TBN, NROW / TBM), 256, 0, stream>>>(
          yg_b, Wout_b, y1, NROW, DM_, DI_, EPI_STORE, 0,
          nullptr, nullptr, nullptr, nullptr, nullptr);
    } else {
      gemm_bf16_wmma<<<dim3(DM_ / TBN, NROW / TBM), 256, 0, stream>>>(
          yg_b, Wout_b, nullptr, NROW, DM_, DI_, EPI_FINAL, 0,
          nullptr, nullptr, y1, input, out);
    }
  }
}
